// SGPBlock_34325378630252
// MI455X (gfx1250) — matrix operand order repaired and checked by reading, not compiled, not run on hardware
//
#include <hip/hip_runtime.h>
#include <hip/hip_bf16.h>

// ---------------------------------------------------------------------------
// SGPBlock fused pipeline for MI455X (gfx1250, wave32, WMMA + TDM)
// B=16, C=512, T=4096, K=3, UP=7, HID=2048, 16 GN groups
// ---------------------------------------------------------------------------

#define B_  16
#define C_  512
#define T_  4096
#define HID_ 2048
#define GRP_ 16
#define CPG_ 32
#define EPS_ 1e-5f

typedef __attribute__((ext_vector_type(16))) __bf16 v16bf;
typedef __attribute__((ext_vector_type(8)))  float  v8f;
typedef __attribute__((ext_vector_type(4)))  unsigned int u32x4;
typedef __attribute__((ext_vector_type(8)))  int    i32x8;
typedef __attribute__((ext_vector_type(4)))  int    i32x4;

#if __has_builtin(__builtin_amdgcn_tensor_load_to_lds) && __has_builtin(__builtin_amdgcn_s_wait_tensorcnt)
#define USE_TDM 1
#else
#define USE_TDM 0
#endif

static __device__ __forceinline__ unsigned short f2bf(float f) {
    unsigned int u = __float_as_uint(f);
    unsigned int r = (u + 0x7FFFu + ((u >> 16) & 1u)) >> 16;  // RNE bf16
    return (unsigned short)r;
}

// --------------------------- zero accumulators -----------------------------
__global__ void kz_zero(float* gnsum) {
    int i = threadIdx.x;
    if (i < B_ * GRP_ * 2) gnsum[i] = 0.0f;
}

// ------------------- pack MLP weights into WMMA A fragments ----------------
// A-matrix 16x32 bf16 layout (ISA 7.12.2): lane<16 -> K {0..7,16..23},
// lane>=16 -> K {8..15,24..31}; each lane's 16 elems stored contiguously.
__global__ void k_pack_w1(const float* __restrict__ w, unsigned short* __restrict__ dst) {
    int tid = blockIdx.x * 256 + threadIdx.x;
    if (tid >= HID_ * C_) return;
    int e    = tid & 15;
    int lane = (tid >> 4) & 31;
    int tile = tid >> 9;            // mt*16 + kt
    int kt = tile & 15;
    int mt = tile >> 4;
    int kl = (lane < 16) ? ((e < 8) ? e : e + 8) : ((e < 8) ? e + 8 : e + 16);
    int m  = mt * 16 + (lane & 15);
    int k  = kt * 32 + kl;
    dst[tid] = f2bf(w[m * C_ + k]);
}

__global__ void k_pack_w2(const float* __restrict__ w, unsigned short* __restrict__ dst) {
    int tid = blockIdx.x * 256 + threadIdx.x;
    if (tid >= C_ * HID_) return;
    int e    = tid & 15;
    int lane = (tid >> 4) & 31;
    int tile = tid >> 9;            // mt*64 + kt
    int kt = tile & 63;
    int mt = tile >> 6;
    int kl = (lane < 16) ? ((e < 8) ? e : e + 8) : ((e < 8) ? e + 8 : e + 16);
    int m  = mt * 16 + (lane & 15);
    int k  = kt * 32 + kl;
    dst[tid] = f2bf(w[m * HID_ + k]);
}

// ----------------------- K1: channel-LN stats per (b,t) --------------------
__global__ __launch_bounds__(256) void k1_lnstats(
    const float* __restrict__ x, float* __restrict__ mu_rs)
{
    int blk = blockIdx.x;               // B * (T/256)
    int b   = blk >> 4;
    int t   = ((blk & 15) << 8) + threadIdx.x;
    const float* xb = x + (size_t)b * C_ * T_;

    float s = 0.0f, s2 = 0.0f;
    for (int c = 0; c < C_; ++c) {
        float v = xb[(size_t)c * T_ + t];
        s += v; s2 += v * v;
    }
    float mu  = s * (1.0f / C_);
    float var = s2 * (1.0f / C_) - mu * mu;
    mu_rs[((size_t)b * T_ + t) * 2]     = mu;
    mu_rs[((size_t)b * T_ + t) * 2 + 1] = rsqrtf(var + EPS_);
}

// ---------------- K1b: temporal pool of LN output, per (b,c) ---------------
// pooled = mean_t(LN(x)) = ln_w * mean_t((x-mu)*rs) + ln_b ; no atomics
__global__ __launch_bounds__(256) void k1b_pool(
    const float* __restrict__ x, const float* __restrict__ mu_rs,
    const float* __restrict__ lnw, const float* __restrict__ lnb,
    float* __restrict__ pooled)
{
    __shared__ float red[8];
    int bc = blockIdx.x;                // B*C
    int b  = bc >> 9, c = bc & (C_ - 1);
    const float* xr = x + (size_t)bc * T_;
    const float* mr = mu_rs + (size_t)b * T_ * 2;
    float s = 0.0f;
    for (int t = threadIdx.x; t < T_; t += 256)
        s += (xr[t] - mr[t * 2]) * mr[t * 2 + 1];
    for (int off = 16; off; off >>= 1) s += __shfl_xor(s, off, 32);
    if ((threadIdx.x & 31) == 0) red[threadIdx.x >> 5] = s;
    __syncthreads();
    if (threadIdx.x == 0) {
        float tot = 0.0f;
        #pragma unroll
        for (int i = 0; i < 8; ++i) tot += red[i];
        pooled[bc] = lnw[c] * tot * (1.0f / T_) + lnb[c];
    }
}

// ------------------------------ K2: gate phi -------------------------------
__global__ void k2_phi(const float* __restrict__ pooled, const float* __restrict__ gw,
                       const float* __restrict__ gb, float* __restrict__ phi) {
    int i = blockIdx.x * 256 + threadIdx.x;
    if (i < B_ * C_) {
        int c = i & (C_ - 1);
        float v = pooled[i] * gw[c] + gb[c];
        phi[i] = v > 0.0f ? v : 0.0f;
    }
}

// ---------- K3: LN recompute + depthwise convs + residual + GN sums --------
__global__ __launch_bounds__(256) void k3_fused(
    const float* __restrict__ x, const float* __restrict__ lnw, const float* __restrict__ lnb,
    const float* __restrict__ psiw, const float* __restrict__ psib,
    const float* __restrict__ fcw,  const float* __restrict__ fcb,
    const float* __restrict__ cww,  const float* __restrict__ cwb,
    const float* __restrict__ kww,  const float* __restrict__ kwb,
    const float* __restrict__ mu_rs, const float* __restrict__ phi,
    float* __restrict__ out, float* __restrict__ gnsum)
{
    __shared__ float lnrow[272];
    __shared__ float muS[272];
    __shared__ float rsS[272];

    int blk = blockIdx.x;               // B * (T/256)
    int b   = blk >> 4;
    int t0  = (blk & 15) << 8;
    int tid = threadIdx.x;
    int t   = t0 + tid;
    int lane = tid & 31;

    for (int i = tid; i < 262; i += 256) {
        int tt = t0 - 3 + i;
        if (tt >= 0 && tt < T_) {
            muS[i] = mu_rs[((size_t)b * T_ + tt) * 2];
            rsS[i] = mu_rs[((size_t)b * T_ + tt) * 2 + 1];
        } else { muS[i] = 0.0f; rsS[i] = 0.0f; }
    }
    __syncthreads();

    float gs1 = 0.0f, gs2 = 0.0f;
    for (int c = 0; c < C_; ++c) {
        float w  = lnw[c], bb = lnb[c];
        const float* xr = x + ((size_t)b * C_ + c) * T_;
        for (int i = tid; i < 262; i += 256) {
            int tt = t0 - 3 + i;
            lnrow[i] = (tt >= 0 && tt < T_) ? ((xr[tt] - muS[i]) * rsS[i] * w + bb) : 0.0f;
        }
        __syncthreads();

        int ci = tid + 3;
        float psi = psib[c] + psiw[c*3]*lnrow[ci-1] + psiw[c*3+1]*lnrow[ci] + psiw[c*3+2]*lnrow[ci+1];
        float cw  = cwb[c]  + cww[c*3]*lnrow[ci-1]  + cww[c*3+1]*lnrow[ci]  + cww[c*3+2]*lnrow[ci+1];
        float ckw = kwb[c];
        #pragma unroll
        for (int j = 0; j < 7; ++j) ckw += kww[c*7+j] * lnrow[ci-3+j];
        float lc  = lnrow[ci];
        float fcv = lc * fcw[c] + fcb[c];
        float xc  = xr[t];
        float ov  = xc + (fcv * phi[b * C_ + c] + (cw + ckw) * psi + lc);
        out[((size_t)b * C_ + c) * T_ + t] = ov;

        gs1 += ov; gs2 += ov * ov;
        if ((c & (CPG_ - 1)) == (CPG_ - 1)) {
            float a = gs1, e = gs2;
            for (int off = 16; off; off >>= 1) { a += __shfl_xor(a, off, 32); e += __shfl_xor(e, off, 32); }
            if (lane == 0) {
                int g = c >> 5;
                atomicAdd(&gnsum[(b * GRP_ + g) * 2],     a);
                atomicAdd(&gnsum[(b * GRP_ + g) * 2 + 1], e);
            }
            gs1 = 0.0f; gs2 = 0.0f;
        }
        __syncthreads();
    }
}

// ------------------------- K4: finalize GN stats ---------------------------
__global__ void k4_gnstats(const float* __restrict__ gnsum, float* __restrict__ gnstats) {
    int i = blockIdx.x * 256 + threadIdx.x;
    if (i < B_ * GRP_) {
        float n  = (float)CPG_ * (float)T_;
        float s  = gnsum[i*2], s2 = gnsum[i*2+1];
        float mu = s / n;
        float var = s2 / n - mu * mu;
        gnstats[i*2]     = mu;
        gnstats[i*2 + 1] = rsqrtf(var + EPS_);
    }
}

// ------------- K5: TDM stage + GN + fused bf16-WMMA MLP + residual ---------
// One block per 64-wide (b,t) tile; 8 waves; N=64 (4 WMMA col-tiles, A reuse x4);
// HID processed in 4 chunks of 512. LDS: 128K raw tile (TDM dest, aliased over
// 64K h-chunk buffer) + 64K act fragments = 192 KB.
__global__ __launch_bounds__(256) void k5_mlp(
    const unsigned short* __restrict__ w1pk, const unsigned short* __restrict__ w2pk,
    const float* __restrict__ b1, const float* __restrict__ b2,
    const float* __restrict__ gnw, const float* __restrict__ gnb,
    const float* __restrict__ gnstats, float* __restrict__ out)
{
    __shared__ __align__(128) char smem[196608];
    float*          rawS = (float*)smem;                      // [512][64] f32 staging
    unsigned short* hS   = (unsigned short*)smem;             // 64 KB (reused after staging)
    unsigned short* actS = (unsigned short*)(smem + 131072);  // 64 KB persistent

    int blk  = blockIdx.x;              // B * (T/64)
    int b    = blk >> 6;
    int t0   = (blk & 63) << 6;
    int tid  = threadIdx.x;
    int lane = tid & 31;
    int wv   = tid >> 5;
    const float* tileBase = out + (size_t)b * C_ * T_ + t0;

#if USE_TDM
    // --- Tensor Data Mover: DMA the [512 x 64] f32 tile (row stride T) to LDS
    if (wv == 0) {
        unsigned long long ga = (unsigned long long)(uintptr_t)tileBase;
        unsigned ldsoff = (unsigned)(uintptr_t)rawS;          // addr[31:0] = LDS offset
        u32x4 g0 = { 1u,                                       // count=1, user desc
                     ldsoff,                                   // lds_addr
                     (unsigned)(ga & 0xffffffffu),             // global_addr[31:0]
                     (unsigned)((ga >> 32) & 0x01ffffffu) | 0x80000000u }; // addr[56:32] | type=2
        i32x8 g1 = { 0x00020000,                               // data_size=4B, no mask/pad
                     (int)(T_ << 16),                          // tensor_dim0[15:0] in [31:16]
                     (int)((T_ >> 16) | (C_ << 16)),           // tensor_dim0[31:16], tensor_dim1[15:0]
                     (int)((C_ >> 16) | (64 << 16)),           // tensor_dim1[31:16], tile_dim0=64
                     (int)C_,                                  // tile_dim1=512, tile_dim2=0
                     (int)T_,                                  // tensor_dim0_stride[31:0]
                     0, 0 };
        i32x4 gz4 = { 0, 0, 0, 0 };
        i32x8 gz8 = { 0, 0, 0, 0, 0, 0, 0, 0 };
        __builtin_amdgcn_tensor_load_to_lds(g0, g1, gz4, gz4, gz8, 0);
        __builtin_amdgcn_s_wait_tensorcnt(0);
    }
    __syncthreads();
#endif

    // Stage: GN-normalize tile -> bf16 B-fragments in LDS
    for (int i = tid; i < C_ * 64; i += 256) {
        int c = i >> 6, col = i & 63;
        int nt = col >> 4, n = col & 15;
        int g = c >> 5;
        float mean = gnstats[(b * GRP_ + g) * 2];
        float rstd = gnstats[(b * GRP_ + g) * 2 + 1];
#if USE_TDM
        float v = rawS[i];
#else
        float v = tileBase[(size_t)c * T_ + col];
#endif
        float a = (v - mean) * rstd * gnw[c] + gnb[c];
        int kin = c & 31;
        int fl  = ((kin >> 3) & 1) * 16 + n;
        actS[((c >> 5) * 4 + nt) * 512 + fl * 16 + (kin & 7) + ((kin >= 16) ? 8 : 0)] = f2bf(a);
    }
    __syncthreads();

    v8f acc2[16] = {};                  // GEMM2: rows [wv*64,+64) x 4 n-tiles
    int n_col = lane & 15, hi = lane >> 4;

    for (int ch = 0; ch < 4; ++ch) {
        // ---- GEMM1: 4 m-tiles x 4 n-tiles of this HID chunk (A reused x4)
        for (int mt = 0; mt < 4; ++mt) {
            int mbase = ch * 512 + wv * 64 + mt * 16;
            int mtile = mbase >> 4;
            v8f acc[4] = {};
            #pragma unroll
            for (int kt = 0; kt < 16; ++kt) {
                const unsigned short* ap = w1pk + ((size_t)(mtile * 16 + kt) * 32 + lane) * 16;
                __builtin_prefetch((const void*)(ap + 512), 0, 0);
                v16bf a = *(const v16bf*)ap;
                #pragma unroll
                for (int nt = 0; nt < 4; ++nt) {
                    v16bf bf = *(const v16bf*)(&actS[(kt * 4 + nt) * 512 + lane * 16]);
                    acc[nt] = __builtin_amdgcn_wmma_f32_16x16x32_bf16(
                                false, a, false, bf, (short)0, acc[nt], false, false);
                }
            }
            // bias + exact GELU + bf16 -> hS (B-frag layout, local chunk rows)
            #pragma unroll
            for (int nt = 0; nt < 4; ++nt) {
                #pragma unroll
                for (int r = 0; r < 8; ++r) {
                    int m  = mbase + r + hi * 8;
                    float v = acc[nt][r] + b1[m];
                    v = 0.5f * v * (1.0f + erff(v * 0.70710678118f));
                    int ml  = m - ch * 512;
                    int kin = ml & 31;
                    int fl  = ((kin >> 3) & 1) * 16 + n_col;
                    hS[((ml >> 5) * 4 + nt) * 512 + fl * 16 + (kin & 7) + ((kin >= 16) ? 8 : 0)] = f2bf(v);
                }
            }
        }
        __syncthreads();

        // ---- GEMM2 partial: accumulate this chunk into persistent tiles
        #pragma unroll
        for (int mt = 0; mt < 4; ++mt) {
            int mbase = wv * 64 + mt * 16;
            int mtile = mbase >> 4;
            #pragma unroll
            for (int kt = 0; kt < 16; ++kt) {
                int ktg = ch * 16 + kt;
                const unsigned short* ap = w2pk + ((size_t)(mtile * 64 + ktg) * 32 + lane) * 16;
                __builtin_prefetch((const void*)(ap + 512), 0, 0);
                v16bf a = *(const v16bf*)ap;
                #pragma unroll
                for (int nt = 0; nt < 4; ++nt) {
                    v16bf bf = *(const v16bf*)(&hS[(kt * 4 + nt) * 512 + lane * 16]);
                    acc2[mt * 4 + nt] = __builtin_amdgcn_wmma_f32_16x16x32_bf16(
                                false, a, false, bf, (short)0, acc2[mt * 4 + nt], false, false);
                }
            }
        }
        __syncthreads();
    }

    // Epilogue: d_out += h + b2 (in place; each element owned by one lane)
    #pragma unroll
    for (int mt = 0; mt < 4; ++mt) {
        int mbase = wv * 64 + mt * 16;
        #pragma unroll
        for (int nt = 0; nt < 4; ++nt) {
            #pragma unroll
            for (int r = 0; r < 8; ++r) {
                int c = mbase + r + hi * 8;
                size_t off = ((size_t)b * C_ + c) * T_ + t0 + nt * 16 + n_col;
                out[off] = out[off] + acc2[mt * 4 + nt][r] + b2[c];
            }
        }
    }
}

// ---------------------------------------------------------------------------
extern "C" void kernel_launch(void* const* d_in, const int* in_sizes, int n_in,
                              void* d_out, int out_size, void* d_ws, size_t ws_size,
                              hipStream_t stream) {
    const float* x      = (const float*)d_in[0];
    const float* ln_w   = (const float*)d_in[1];
    const float* ln_b   = (const float*)d_in[2];
    const float* psi_w  = (const float*)d_in[3];
    const float* psi_b  = (const float*)d_in[4];
    const float* fc_w   = (const float*)d_in[5];
    const float* fc_b   = (const float*)d_in[6];
    const float* cw_w   = (const float*)d_in[7];
    const float* cw_b   = (const float*)d_in[8];
    const float* kw_w   = (const float*)d_in[9];
    const float* kw_b   = (const float*)d_in[10];
    const float* gfc_w  = (const float*)d_in[11];
    const float* gfc_b  = (const float*)d_in[12];
    const float* gn_w   = (const float*)d_in[13];
    const float* gn_b   = (const float*)d_in[14];
    const float* mlp_w1 = (const float*)d_in[15];
    const float* mlp_b1 = (const float*)d_in[16];
    const float* mlp_w2 = (const float*)d_in[17];
    const float* mlp_b2 = (const float*)d_in[18];
    float* out = (float*)d_out;

    // workspace layout (all offsets 32B-aligned)
    char* ws = (char*)d_ws;
    float* mu_rs   = (float*)(ws + 0);          // B*T*2 f32   = 512 KiB
    float* pooled  = (float*)(ws + 524288);     // B*C  f32    = 32 KiB
    float* phi     = (float*)(ws + 557056);     // B*C  f32    = 32 KiB
    float* gnsum   = (float*)(ws + 589824);     // B*16*2 f32  = 2 KiB
    float* gnstats = (float*)(ws + 591872);     // B*16*2 f32  = 2 KiB
    unsigned short* w1pk = (unsigned short*)(ws + 593920);   // 2 MiB bf16
    unsigned short* w2pk = (unsigned short*)(ws + 2691072);  // 2 MiB bf16

    kz_zero<<<1, 512, 0, stream>>>(gnsum);

    k_pack_w1<<<(HID_ * C_) / 256, 256, 0, stream>>>(mlp_w1, w1pk);
    k_pack_w2<<<(C_ * HID_) / 256, 256, 0, stream>>>(mlp_w2, w2pk);

    k1_lnstats<<<B_ * (T_ / 256), 256, 0, stream>>>(x, mu_rs);

    k1b_pool<<<B_ * C_, 256, 0, stream>>>(x, mu_rs, ln_w, ln_b, pooled);

    k2_phi<<<(B_*C_ + 255) / 256, 256, 0, stream>>>(pooled, gfc_w, gfc_b, phi);

    k3_fused<<<B_ * (T_ / 256), 256, 0, stream>>>(
        x, ln_w, ln_b, psi_w, psi_b, fc_w, fc_b, cw_w, cw_b, kw_w, kw_b,
        mu_rs, phi, out, gnsum);

    k4_gnstats<<<1, 256, 0, stream>>>(gnsum, gnstats);

    k5_mlp<<<B_ * (T_ / 64), 256, 0, stream>>>(
        w1pk, w2pk, mlp_b1, mlp_b2, gn_w, gn_b, gnstats, out);

    (void)in_sizes; (void)n_in; (void)out_size; (void)ws_size;
}
